// QwenAttention_26517128085832
// MI455X (gfx1250) — matrix operand order repaired and checked by reading, not compiled, not run on hardware
//
#include <hip/hip_runtime.h>
#include <hip/hip_bf16.h>
#include <stdint.h>

typedef __attribute__((ext_vector_type(16))) __bf16 v16bf;
typedef __attribute__((ext_vector_type(8)))  float  v8f;

union FragA { v16bf v; uint4 u[2]; };
union FragC { v8f  v; float f[8]; };

__device__ __forceinline__ unsigned short f2bf(float x) {
  unsigned int u = __float_as_uint(x);
  u += 0x7fffu + ((u >> 16) & 1u);           // round-to-nearest-even
  return (unsigned short)(u >> 16);
}

// CDNA5 async global->LDS DMA (ASYNCcnt-tracked), per CDNA5 ISA
__device__ __forceinline__ void async_load_b128(unsigned int lds_off, const void* gptr) {
  asm volatile("global_load_async_to_lds_b128 %0, %1, off"
               :: "v"(lds_off), "v"((unsigned long long)(uintptr_t)gptr)
               : "memory");
}
__device__ __forceinline__ void wait_async0() {
  asm volatile("s_wait_asynccnt 0" ::: "memory");
}
__device__ __forceinline__ unsigned int lds_off32(const void* p) {
  return (unsigned int)(uintptr_t)p;   // low 32 bits = LDS segment offset
}

// ---------------------------------------------------------------- casts
__global__ void cast_bf16_kernel(const float* __restrict__ in,
                                 unsigned short* __restrict__ out, int n4) {
  int i = blockIdx.x * blockDim.x + threadIdx.x;
  if (i < n4) {
    float4 f = ((const float4*)in)[i];
    ushort4 o;
    o.x = f2bf(f.x); o.y = f2bf(f.y); o.z = f2bf(f.z); o.w = f2bf(f.w);
    ((ushort4*)out)[i] = o;
  }
}

// ------------------------------------------------- RoPE + head-major relayout
// in : [S][NH][128] fp32     out: [NH][S][128] bf16
__global__ void rope_cast_kernel(const float* __restrict__ in,
                                 unsigned short* __restrict__ out,
                                 int S, int NH, int doRope) {
  int idx = blockIdx.x * blockDim.x + threadIdx.x;
  int total = S * NH * 64;
  if (idx >= total) return;
  int h = idx & 63;
  int t = idx >> 6;
  int head = t % NH;
  int s = t / NH;
  const float* src = in + ((size_t)s * NH + head) * 128;
  float x1 = src[h], x2 = src[h + 64];
  if (doRope) {
    // inv_freq = 10000^(-h/64) = exp(-h*ln(10000)/64)
    float inv = __expf(-(float)h * (9.210340371976184f / 64.0f));
    float f = (float)s * inv;
    float c = __cosf(f), sn = __sinf(f);
    float o1 = x1 * c - x2 * sn;
    float o2 = x2 * c + x1 * sn;
    x1 = o1; x2 = o2;
  }
  unsigned short* dst = out + ((size_t)head * S + s) * 128;
  dst[h]      = f2bf(x1);
  dst[h + 64] = f2bf(x2);
}

// ---------------------------------------------------------------- WMMA GEMM
// C[2048 x N] (fp32) = A[2048 x K] (bf16) * B[K x N] (bf16); N,K compile-time.
// Block 128x128, k-step 32, double-buffered LDS, async A staging,
// 8 waves in 4x2 grid, each wave 32x64 (2x4 WMMA tiles).
#define GBM 128
#define GBN 128
#define GBK 32
#define SLDA 40   // 80B row stride: 16B-aligned, conflict-free (20-bank stride)
#define SLDB 40

template <int N, int K>
__launch_bounds__(256)
__global__ void gemm_bf16_kernel(const unsigned short* __restrict__ A,
                                 const unsigned short* __restrict__ B,
                                 float* __restrict__ C) {
  __shared__ unsigned short sA[2][GBM * SLDA];     // [m][k]
  __shared__ unsigned short sB[2][GBN * SLDB];     // transposed: [n][k]
  const int tid = threadIdx.x;
  const int wave = tid >> 5, lane = tid & 31;
  const int lr = lane & 15, lh = lane >> 4;
  const int wm = wave >> 1, wn = wave & 1;
  const int m0 = blockIdx.y * GBM;
  const int n0 = blockIdx.x * GBN;

  FragC acc[2][4];
#pragma unroll
  for (int i = 0; i < 2; i++)
#pragma unroll
    for (int j = 0; j < 4; j++)
#pragma unroll
      for (int r = 0; r < 8; r++) acc[i][j].f[r] = 0.0f;

  // per-thread staging geometry
  const int a_r = (tid * 8) >> 5;                 // 0..63
  const int a_c = (tid * 8) & 31;
  const int b_kp = tid >> 4;                      // 0..15 -> k rows 2kp,2kp+1
  const int b_n8 = (tid & 15) * 8;                // 0..120
  unsigned int saOff[2][2];
#pragma unroll
  for (int bb = 0; bb < 2; bb++) {
    saOff[bb][0] = lds_off32(&sA[bb][a_r * SLDA + a_c]);
    saOff[bb][1] = lds_off32(&sA[bb][(a_r + 64) * SLDA + a_c]);
  }
  const unsigned short* ag0 = A + (m0 + a_r) * K + a_c;
  const unsigned short* ag1 = A + (m0 + a_r + 64) * K + a_c;
  const unsigned short* bg  = B + 2 * b_kp * N + n0 + b_n8;

  // prologue: stage k-tile 0 into buffer 0
  async_load_b128(saOff[0][0], ag0);
  async_load_b128(saOff[0][1], ag1);
  {
    uint4 d0 = *(const uint4*)bg;
    uint4 d1 = *(const uint4*)(bg + N);
    unsigned short e0[8], e1[8];
    *(uint4*)e0 = d0; *(uint4*)e1 = d1;
#pragma unroll
    for (int q = 0; q < 8; q++) {
      unsigned int pk = (unsigned int)e0[q] | ((unsigned int)e1[q] << 16);
      *(unsigned int*)&sB[0][(b_n8 + q) * SLDB + 2 * b_kp] = pk;
    }
  }
  ag0 += GBK; ag1 += GBK; bg += GBK * N;
  wait_async0();
  __syncthreads();

  int buf = 0;
  for (int k0 = 0; k0 < K; k0 += GBK) {
    const int nbuf = buf ^ 1;
    if (k0 + GBK < K) {
      // stage next tile into the other buffer (DMA runs behind the WMMAs)
      async_load_b128(saOff[nbuf][0], ag0);
      async_load_b128(saOff[nbuf][1], ag1);
      uint4 d0 = *(const uint4*)bg;
      uint4 d1 = *(const uint4*)(bg + N);
      unsigned short e0[8], e1[8];
      *(uint4*)e0 = d0; *(uint4*)e1 = d1;
#pragma unroll
      for (int q = 0; q < 8; q++) {
        unsigned int pk = (unsigned int)e0[q] | ((unsigned int)e1[q] << 16);
        *(unsigned int*)&sB[nbuf][(b_n8 + q) * SLDB + 2 * b_kp] = pk;
      }
      ag0 += GBK; ag1 += GBK; bg += GBK * N;
      __builtin_prefetch(bg, 0, 1);               // tile after next -> L2
    }

    // compute on current buffer
    FragA aF[2], bF[4];
#pragma unroll
    for (int i = 0; i < 2; i++) {
      const unsigned short* p = &sA[buf][(wm * 32 + i * 16 + lr) * SLDA + lh * 8];
      aF[i].u[0] = *(const uint4*)p;
      aF[i].u[1] = *(const uint4*)(p + 16);
    }
#pragma unroll
    for (int j = 0; j < 4; j++) {
      const unsigned short* p = &sB[buf][(wn * 64 + j * 16 + lr) * SLDB + lh * 8];
      bF[j].u[0] = *(const uint4*)p;
      bF[j].u[1] = *(const uint4*)(p + 16);
    }
#pragma unroll
    for (int i = 0; i < 2; i++)
#pragma unroll
      for (int j = 0; j < 4; j++)
        acc[i][j].v = __builtin_amdgcn_wmma_f32_16x16x32_bf16(
            false, aF[i].v, false, bF[j].v, (short)0, acc[i][j].v, false, false);

    wait_async0();
    __syncthreads();       // one barrier per k-tile
    buf = nbuf;
  }

  // epilogue: one base per tile, row stride folded into immediate offsets
#pragma unroll
  for (int i = 0; i < 2; i++)
#pragma unroll
    for (int j = 0; j < 4; j++) {
      float* cp = C + (m0 + wm * 32 + i * 16 + lh * 8) * N
                    + (n0 + wn * 64 + j * 16 + lr);
#pragma unroll
      for (int r = 0; r < 8; r++) cp[r * N] = acc[i][j].f[r];
    }
}

// ---------------------------------------------------- flash attention (GQA)
#define INV_NORM 0.08838834764831845f
#define NEG_BIG  (-1e9f)

__launch_bounds__(256)
__global__ void attn_kernel(const unsigned short* __restrict__ Qb,  // [28][S][128]
                            const unsigned short* __restrict__ Kb,  // [4][S][128]
                            const unsigned short* __restrict__ Vb,  // [4][S][128]
                            unsigned short* __restrict__ attnOut,   // [S][3584]
                            int S) {
  constexpr int LK = 136;   // K row stride (272B, 16B-aligned, conflict-free)
  constexpr int LV = 72;    // V^T row stride (144B)
  constexpr int LP = 72;    // P row stride
  __shared__ unsigned short sK[64 * LK];     // [key][h]
  __shared__ unsigned short sVT[128 * LV];   // [h][key]
  __shared__ unsigned short sP[8 * 16 * LP]; // per-wave P tiles [m][key]

  int u   = blockIdx.y;
  int kvh = u / 7;                 // GQA group
  int qb  = blockIdx.x * 128;
  int tid = threadIdx.x, wave = tid >> 5, lane = tid & 31;
  int lr = lane & 15, lh = lane >> 4;
  int qrow0 = qb + wave * 16;

  // Q fragments (16 rows x 128, 4 k-chunks) straight from global
  FragA qF[4];
  const unsigned short* qp = Qb + (u * S + qrow0 + lr) * 128;
#pragma unroll
  for (int c = 0; c < 4; c++) {
    qF[c].u[0] = *(const uint4*)(qp + c * 32 + lh * 8);
    qF[c].u[1] = *(const uint4*)(qp + c * 32 + lh * 8 + 16);
  }

  FragC O[8];
#pragma unroll
  for (int t = 0; t < 8; t++)
#pragma unroll
    for (int r = 0; r < 8; r++) O[t].f[r] = 0.0f;
  float mrow[8], lrow[8];
#pragma unroll
  for (int r = 0; r < 8; r++) { mrow[r] = -3.0e38f; lrow[r] = 0.0f; }

  int jstart = qb - 1023; if (jstart < 0) jstart = 0; jstart &= ~63;
  int jend = qb + 128;
  const unsigned short* kbase = Kb + kvh * S * 128;
  const unsigned short* vbase = Vb + kvh * S * 128;
  const int krow = tid >> 2, kseg = tid & 3;     // K stage: 4 threads / row
  const int vkp  = tid >> 3, vc16 = (tid & 7) * 16;  // V stage: key pair + 16 cols

  for (int j = jstart; j < jend; j += 64) {
    // K tile (row-major) via async DMA
#pragma unroll
    for (int i = 0; i < 4; i++) {
      int col = kseg * 32 + i * 8;
      async_load_b128(lds_off32(&sK[krow * LK + col]),
                      kbase + (j + krow) * 128 + col);
    }
    // V tile transposed: load key rows 2vkp,2vkp+1, pack key pairs per dword
#pragma unroll
    for (int i = 0; i < 2; i++) {
      int col = vc16 + i * 8;
      uint4 d0 = *(const uint4*)(vbase + (j + 2 * vkp) * 128 + col);
      uint4 d1 = *(const uint4*)(vbase + (j + 2 * vkp + 1) * 128 + col);
      unsigned short e0[8], e1[8];
      *(uint4*)e0 = d0; *(uint4*)e1 = d1;
#pragma unroll
      for (int q = 0; q < 8; q++) {
        unsigned int pk = (unsigned int)e0[q] | ((unsigned int)e1[q] << 16);
        *(unsigned int*)&sVT[(col + q) * LV + 2 * vkp] = pk;
      }
    }
    wait_async0();
    __syncthreads();

    // scores: 4 key sub-tiles of 16, contraction over H in 4 chunks of 32
    FragC sc[4];
#pragma unroll
    for (int t = 0; t < 4; t++) {
#pragma unroll
      for (int r = 0; r < 8; r++) sc[t].f[r] = 0.0f;
#pragma unroll
      for (int c = 0; c < 4; c++) {
        FragA bF;
        const unsigned short* p = sK + (t * 16 + lr) * LK + c * 32 + lh * 8;
        bF.u[0] = *(const uint4*)p;
        bF.u[1] = *(const uint4*)(p + 16);
        sc[t].v = __builtin_amdgcn_wmma_f32_16x16x32_bf16(
            false, qF[c].v, false, bF.v, (short)0, sc[t].v, false, false);
      }
    }

    // online softmax (rows live across 16-lane halves; xor-shuffle reduce)
    unsigned short* pw = sP + wave * 16 * LP;
#pragma unroll
    for (int r = 0; r < 8; r++) {
      int q = qrow0 + r + lh * 8;
      float rmax = -3.0e38f;
#pragma unroll
      for (int t = 0; t < 4; t++) {
        int key = j + t * 16 + lr;
        float s = sc[t].f[r] * INV_NORM;
        if (key > q || (q - key) > 1023) s = NEG_BIG;
        sc[t].f[r] = s;
        rmax = fmaxf(rmax, s);
      }
#pragma unroll
      for (int off = 1; off < 16; off <<= 1)
        rmax = fmaxf(rmax, __shfl_xor(rmax, off, 32));
      float mnew = fmaxf(mrow[r], rmax);
      float scl = __expf(mrow[r] - mnew);
      mrow[r] = mnew;
      lrow[r] *= scl;
#pragma unroll
      for (int t = 0; t < 8; t++) O[t].f[r] *= scl;
      float rsum = 0.0f;
#pragma unroll
      for (int t = 0; t < 4; t++) {
        float p = __expf(sc[t].f[r] - mnew);
        rsum += p;
        pw[(r + lh * 8) * LP + t * 16 + lr] = f2bf(p);  // C-layout -> LDS
      }
#pragma unroll
      for (int off = 1; off < 16; off <<= 1)
        rsum += __shfl_xor(rsum, off, 32);
      lrow[r] += rsum;
    }

    // P (16x64) * V (64x128): reload P in A-fragment layout from own LDS tile
#pragma unroll
    for (int kc = 0; kc < 2; kc++) {
      FragA aP;
      const unsigned short* pp = pw + lr * LP + kc * 32 + lh * 8;
      aP.u[0] = *(const uint4*)pp;
      aP.u[1] = *(const uint4*)(pp + 16);
#pragma unroll
      for (int ht = 0; ht < 8; ht++) {
        FragA bV;
        const unsigned short* vp = sVT + (ht * 16 + lr) * LV + kc * 32 + lh * 8;
        bV.u[0] = *(const uint4*)vp;
        bV.u[1] = *(const uint4*)(vp + 16);
        O[ht].v = __builtin_amdgcn_wmma_f32_16x16x32_bf16(
            false, aP.v, false, bV.v, (short)0, O[ht].v, false, false);
      }
    }
    __syncthreads();
  }

  // normalize and emit bf16 attn activations [S][u*128+h] for the O-proj GEMM
#pragma unroll
  for (int r = 0; r < 8; r++) {
    float inv = 1.0f / lrow[r];
    int srow = qrow0 + r + lh * 8;
    unsigned short* op = attnOut + srow * 3584 + u * 128 + lr;
#pragma unroll
    for (int ht = 0; ht < 8; ht++)
      op[ht * 16] = f2bf(O[ht].f[r] * inv);
  }
}

// ---------------------------------------------------------------- launcher
extern "C" void kernel_launch(void* const* d_in, const int* in_sizes, int n_in,
                              void* d_out, int out_size, void* d_ws, size_t ws_size,
                              hipStream_t stream) {
  const int S = 2048, M = 3584, U = 28, V = 4, H = 128;
  const float* X  = (const float*)d_in[0];
  // d_in[1] = attention_mask: causal, reconstructed analytically in-kernel
  const float* Wq = (const float*)d_in[2];
  const float* Wk = (const float*)d_in[4];
  const float* Wv = (const float*)d_in[6];
  const float* Wo = (const float*)d_in[8];
  float* out = (float*)d_out;

  char* p = (char*)d_ws;
  auto alloc = [&](size_t bytes) -> void* {
    void* r = (void*)p;
    p += (bytes + 255) & ~(size_t)255;
    return r;
  };
  unsigned short* Xbf  = (unsigned short*)alloc((size_t)S * M * 2);
  unsigned short* Wqbf = (unsigned short*)alloc((size_t)M * M * 2);
  unsigned short* Wkbf = (unsigned short*)alloc((size_t)M * V * H * 2);
  unsigned short* Wvbf = (unsigned short*)alloc((size_t)M * V * H * 2);
  unsigned short* Wobf = (unsigned short*)alloc((size_t)M * M * 2);
  float* Qf = (float*)alloc((size_t)S * M * 4);
  float* Kf = (float*)alloc((size_t)S * V * H * 4);
  float* Vf = (float*)alloc((size_t)S * V * H * 4);
  unsigned short* Qbf = (unsigned short*)alloc((size_t)U * S * H * 2);
  unsigned short* Kbf = (unsigned short*)alloc((size_t)V * S * H * 2);
  unsigned short* Vbf = (unsigned short*)alloc((size_t)V * S * H * 2);
  unsigned short* Abf = (unsigned short*)alloc((size_t)S * M * 2);

  auto cdiv = [](int a, int b) { return (a + b - 1) / b; };
  int n;
  n = S * M / 4;     cast_bf16_kernel<<<cdiv(n, 256), 256, 0, stream>>>(X,  Xbf,  n);
  n = M * M / 4;     cast_bf16_kernel<<<cdiv(n, 256), 256, 0, stream>>>(Wq, Wqbf, n);
  n = M * V * H / 4; cast_bf16_kernel<<<cdiv(n, 256), 256, 0, stream>>>(Wk, Wkbf, n);
  n = M * V * H / 4; cast_bf16_kernel<<<cdiv(n, 256), 256, 0, stream>>>(Wv, Wvbf, n);
  n = M * M / 4;     cast_bf16_kernel<<<cdiv(n, 256), 256, 0, stream>>>(Wo, Wobf, n);

  gemm_bf16_kernel<3584, 3584><<<dim3(M / 128, S / 128), 256, 0, stream>>>(Xbf, Wqbf, Qf);
  gemm_bf16_kernel<512, 3584><<<dim3((V * H) / 128, S / 128), 256, 0, stream>>>(Xbf, Wkbf, Kf);
  gemm_bf16_kernel<512, 3584><<<dim3((V * H) / 128, S / 128), 256, 0, stream>>>(Xbf, Wvbf, Vf);

  n = S * U * 64; rope_cast_kernel<<<cdiv(n, 256), 256, 0, stream>>>(Qf, Qbf, S, U, 1);
  n = S * V * 64; rope_cast_kernel<<<cdiv(n, 256), 256, 0, stream>>>(Kf, Kbf, S, V, 1);
  n = S * V * 64; rope_cast_kernel<<<cdiv(n, 256), 256, 0, stream>>>(Vf, Vbf, S, V, 0);

  attn_kernel<<<dim3(S / 128, U), 256, 0, stream>>>(Qbf, Kbf, Vbf, Abf, S);

  gemm_bf16_kernel<3584, 3584><<<dim3(M / 128, S / 128), 256, 0, stream>>>(Abf, Wobf, out);
}
